// MultiAttention_74277164417663
// MI455X (gfx1250) — matrix operand order repaired and checked by reading, not compiled, not run on hardware
//
#include <hip/hip_runtime.h>

// ---------------------------------------------------------------------------
// Multi-head attention for MI455X (gfx1250), all matmuls via
// V_WMMA_F32_16X16X32_F16 (f16 inputs, f32 accumulate), wave32.
//   B=4, S=2048, D=512, H=8, DK=DV=64
// Pipeline:
//   1) fp32 -> fp16 conversion; weights transposed so every GEMM B-operand is
//      "B^T row-major" (contiguous B-fragment loads)
//   2) QKV projection GEMMs (V written transposed [B,H,DV,S])
//   3) flash-attention: 4 waves/block share one (b,h); K/V 32-key tiles are
//      staged into LDS once per block via the Tensor Data Mover
//      (tensor_load_to_lds, double-buffered, s_wait_tensorcnt + barrier),
//      cutting L2 read traffic 4x. Online softmax via lane shuffles; P routed
//      through LDS to convert C-layout -> A-layout.
//   4) output projection GEMM + bias -> fp32 d_out
// ---------------------------------------------------------------------------

typedef _Float16 half_t;
typedef __attribute__((ext_vector_type(16))) _Float16 v16h;
typedef __attribute__((ext_vector_type(8)))  _Float16 v8h;
typedef __attribute__((ext_vector_type(8)))  float    v8f;
typedef __attribute__((ext_vector_type(4)))  unsigned int v4u;
typedef __attribute__((ext_vector_type(8)))  int v8i;
typedef __attribute__((ext_vector_type(4)))  int v4i;

#define WMMA(a, b, c) \
  __builtin_amdgcn_wmma_f32_16x16x32_f16(false, (a), false, (b), (short)0, (c), false, false)

#if defined(__has_builtin)
#if __has_builtin(__builtin_amdgcn_tensor_load_to_lds) && \
    __has_builtin(__builtin_amdgcn_s_wait_tensorcnt)
#define HAVE_TDM 1
#endif
#endif
#ifndef HAVE_TDM
#define HAVE_TDM 0
#endif

constexpr int Bdim = 4, Sdim = 2048, Ddim = 512, Hh = 8, DKc = 64, DVc = 64;

// A-fragment (16x32 f16, M x K) from row-major matrix, row stride ld elements.
// Lane L: row = L%16, K = (L/16)*16 + {0..15} contiguous.
__device__ inline v16h load_A_frag(const half_t* p, size_t ld, int lane) {
  const int m  = lane & 15;
  const int kb = (lane >> 4) * 16;
  const half_t* r = p + (size_t)m * ld + kb;
  v8h lo = *(const v8h*)(r);        // K = kb .. kb+7
  v8h hi = *(const v8h*)(r + 8);    // K = kb+8 .. kb+15
  v16h a;
#pragma unroll
  for (int i = 0; i < 8; ++i) { a[i] = lo[i]; a[8 + i] = hi[i]; }
  return a;
}

// B-fragment (32x16 f16, K x N) loaded from B^T stored row-major:
// p points at Bt[n0][k0], row stride ld. Lane L: col n = L%16,
// K = (L/16)*16 + {0..15} contiguous.
__device__ inline v16h load_Bt_frag(const half_t* p, size_t ld, int lane) {
  const int n  = lane & 15;
  const int kb = (lane >> 4) * 16;
  const half_t* r = p + (size_t)n * ld + kb;
  v8h lo = *(const v8h*)(r);        // K = kb .. kb+7
  v8h hi = *(const v8h*)(r + 8);    // K = kb+8 .. kb+15
  v16h b;
#pragma unroll
  for (int i = 0; i < 8; ++i) { b[i] = lo[i]; b[8 + i] = hi[i]; }
  return b;
}

#if HAVE_TDM
// Issue one 2D TDM tile load: tile_d1 rows x tile_d0 elements (f16),
// global row stride stride0 (elements), packed contiguously into LDS.
// D# bitfield packing per CDNA5 ISA 8.3/8.4 (count=1, type=2, data_size=2B).
__device__ inline void tdm_load_2d(unsigned lds_addr, const half_t* gptr,
                                   unsigned tile_d0, unsigned tile_d1,
                                   unsigned long long tensor_d0,
                                   unsigned long long tensor_d1,
                                   unsigned long long stride0) {
  unsigned long long ga = (unsigned long long)(size_t)(const void*)gptr;
  v4u g0;
  g0[0] = 1u;                                   // count=1, user mode
  g0[1] = lds_addr;                             // lds_addr [63:32]
  g0[2] = (unsigned)(ga & 0xFFFFFFFFu);         // global_addr [95:64]
  g0[3] = (unsigned)((ga >> 32) & 0x1FFFFFFu)   // global_addr [120:96]
          | (2u << 30);                         // type=2 ("image") [127:126]
  v8i g1;
  g1[0] = (int)(1u << 16);                      // data_size=1 (2 bytes)
  g1[1] = (int)((tensor_d0 & 0xFFFFu) << 16);   // tensor_dim0[15:0] @ [63:48]
  g1[2] = (int)(((tensor_d0 >> 16) & 0xFFFFu) | ((tensor_d1 & 0xFFFFu) << 16));
  g1[3] = (int)(((tensor_d1 >> 16) & 0xFFFFu) | ((tile_d0 & 0xFFFFu) << 16));
  g1[4] = (int)(tile_d1 & 0xFFFFu);             // tile_dim1; tile_dim2=0 (2D)
  g1[5] = (int)(stride0 & 0xFFFFFFFFull);       // tensor_dim0_stride[31:0]
  g1[6] = (int)((stride0 >> 32) & 0xFFFFull);   // stride0[47:32]; stride1=0
  g1[7] = 0;
  v4i z4 = {};
  v8i z8 = {};
  // 6-arg toolchain form: (g0, g1, g2, g3, extra, cpol)
  __builtin_amdgcn_tensor_load_to_lds(g0, g1, z4, z4, z8, 0);
}
#endif

// ------------------------- conversion kernels ------------------------------

__global__ void k_f32_to_f16(const float* __restrict__ in, half_t* __restrict__ out, int n) {
  int i = blockIdx.x * blockDim.x + threadIdx.x;
  if (i < n) out[i] = (half_t)in[i];
}

// W[h][k][n] (f32) -> Wt[h][n][k] (f16): B^T row-major per head.
__global__ void k_transpose_to_f16(const float* __restrict__ W, half_t* __restrict__ Wt,
                                   int Hn, int Kd, int Nd) {
  int i = blockIdx.x * blockDim.x + threadIdx.x;
  int total = Hn * Kd * Nd;
  if (i >= total) return;
  int h = i / (Kd * Nd);
  int rem = i % (Kd * Nd);
  int k = rem / Nd;
  int n = rem % Nd;
  Wt[((size_t)h * Nd + n) * Kd + k] = (half_t)W[i];
}

// ------------------------- QKV projection GEMM -----------------------------
// One wave computes a 16 x 64 output tile for one (b, h, row-tile).
__global__ __launch_bounds__(128) void k_proj_gemm(
    const half_t* __restrict__ X, const half_t* __restrict__ Wt,
    const float* __restrict__ bias, half_t* __restrict__ Out, int transposed) {
  const int lane = threadIdx.x & 31;
  const int wave = threadIdx.x >> 5;
  const int gw   = blockIdx.x * (blockDim.x >> 5) + wave;
  const int qt = gw % (Sdim / 16);
  const int h  = (gw / (Sdim / 16)) % Hh;
  const int b  = gw / ((Sdim / 16) * Hh);

  const half_t* Xp = X + ((size_t)b * Sdim + (size_t)qt * 16) * Ddim;
  const half_t* Wp = Wt + (size_t)h * DKc * Ddim;

  v8f acc[4] = {};
  for (int ks = 0; ks < Ddim; ks += 32) {
    v16h a = load_A_frag(Xp + ks, Ddim, lane);
#pragma unroll
    for (int nt = 0; nt < 4; ++nt) {
      v16h bf = load_Bt_frag(Wp + (size_t)(nt * 16) * Ddim + ks, Ddim, lane);
      acc[nt] = WMMA(a, bf, acc[nt]);
    }
  }

  const int col  = lane & 15;
  const int row0 = (lane >> 4) * 8;
#pragma unroll
  for (int nt = 0; nt < 4; ++nt) {
    float bv = bias[h * DKc + nt * 16 + col];
#pragma unroll
    for (int r = 0; r < 8; ++r) {
      float v = acc[nt][r] + bv;
      int srow = qt * 16 + row0 + r;
      int ncol = nt * 16 + col;
      size_t idx = transposed
          ? (((size_t)b * Hh + h) * DVc + ncol) * Sdim + srow
          : (((size_t)b * Hh + h) * Sdim + srow) * DKc + ncol;
      Out[idx] = (half_t)v;
    }
  }
}

// ------------------------- flash attention ---------------------------------
// Block = 4 waves, all on the same (b,h), consecutive 16-query tiles.
// Per 32-key step: K tile (32x64 f16) and V tile (64x32 f16) staged into LDS
// once per block by the TDM (double-buffered), then each wave runs
// 4 QK^T WMMAs -> online softmax -> 4 PV WMMAs.
__global__ __launch_bounds__(128) void k_attention(
    const half_t* __restrict__ Q,   // [B,H,S,DK]
    const half_t* __restrict__ K,   // [B,H,S,DK]
    const half_t* __restrict__ Vt,  // [B,H,DV,S]
    half_t* __restrict__ Cc) {      // [B,S,H*DV]
  __shared__ half_t ldsP[4 * 16 * 32];
#if HAVE_TDM
  __shared__ half_t ldsK[2 * 32 * 64];
  __shared__ half_t ldsV[2 * 64 * 32];
#endif
  const int lane = threadIdx.x & 31;
  const int wave = threadIdx.x >> 5;
  // Block-uniform (b,h); per-wave query tile.
  const int bh = blockIdx.x / 32;               // 0..31
  const int qt = (blockIdx.x % 32) * 4 + wave;  // 0..127
  const int h  = bh % Hh;
  const int b  = bh / Hh;

  const half_t* Qp = Q + (((size_t)b * Hh + h) * Sdim + (size_t)qt * 16) * DKc;
  const half_t* Kp = K + ((size_t)b * Hh + h) * Sdim * DKc;
  const half_t* Vp = Vt + ((size_t)b * Hh + h) * DVc * Sdim;
  half_t* Pl = ldsP + wave * 16 * 32;

  const v16h aQ0 = load_A_frag(Qp, DKc, lane);        // d = 0..31
  const v16h aQ1 = load_A_frag(Qp + 32, DKc, lane);   // d = 32..63

  float m8[8], l8[8];
  v8f O[4] = {};
#pragma unroll
  for (int r = 0; r < 8; ++r) { m8[r] = -1e30f; l8[r] = 0.f; }

  const float scale = 0.125f;  // 1/sqrt(DK)
  const int col  = lane & 15;
  const int row0 = (lane >> 4) * 8;
  const int nIter = Sdim / 32;

#if HAVE_TDM
  if (wave == 0) {  // prologue: stage tiles for kb=0 into buffer 0
    tdm_load_2d((unsigned)(size_t)(const void*)(ldsK), Kp, 64, 32, 64, Sdim, 64);
    tdm_load_2d((unsigned)(size_t)(const void*)(ldsV), Vp, 32, 64, Sdim, 64, Sdim);
  }
#endif

  for (int it = 0; it < nIter; ++it) {
    const int kb = it * 32;
#if HAVE_TDM
    const int cur = it & 1;
    if (wave == 0) {
      if (it + 1 < nIter) {  // stage next tiles into the other buffer
        tdm_load_2d((unsigned)(size_t)(const void*)(ldsK + (cur ^ 1) * 32 * 64),
                    Kp + (size_t)(kb + 32) * DKc, 64, 32, 64, Sdim, 64);
        tdm_load_2d((unsigned)(size_t)(const void*)(ldsV + (cur ^ 1) * 64 * 32),
                    Vp + (kb + 32), 32, 64, Sdim, 64, Sdim);
        __builtin_amdgcn_s_wait_tensorcnt(2);  // in-order: current pair landed
      } else {
        __builtin_amdgcn_s_wait_tensorcnt(0);
      }
    }
    __syncthreads();  // current K/V tiles visible to all 4 waves
    const half_t* Kt = ldsK + cur * 32 * 64;   // [32 keys][64 d], ld=64
    const half_t* Vtl = ldsV + cur * 64 * 32;  // [64 dv][32 s], ld=32
#else
    __builtin_prefetch(Kp + (size_t)(kb + 32) * DKc, 0, 1);  // global_prefetch_b8
    const half_t* Kt = Kp + (size_t)kb * DKc;
    const half_t* Vtl = Vp;  // indexed with absolute kb below
#endif

    v8f s0 = {}, s1 = {};
    {
      v16h b00 = load_Bt_frag(Kt,                 DKc, lane);  // keys 0..15, d 0..31
      v16h b01 = load_Bt_frag(Kt + 32,            DKc, lane);  // keys 0..15, d 32..63
      s0 = WMMA(aQ0, b00, s0);
      s0 = WMMA(aQ1, b01, s0);
      v16h b10 = load_Bt_frag(Kt + 16 * DKc,      DKc, lane);  // keys 16..31
      v16h b11 = load_Bt_frag(Kt + 16 * DKc + 32, DKc, lane);
      s1 = WMMA(aQ0, b10, s1);
      s1 = WMMA(aQ1, b11, s1);
    }

    float tm[8], al[8], p0[8], p1[8], rs[8];
#pragma unroll
    for (int r = 0; r < 8; ++r) {
      s0[r] *= scale;
      s1[r] *= scale;
      tm[r] = fmaxf(s0[r], s1[r]);
    }
#pragma unroll
    for (int r = 0; r < 8; ++r) {  // row max across the 16 lanes of each half
      tm[r] = fmaxf(tm[r], __shfl_xor(tm[r], 1, 32));
      tm[r] = fmaxf(tm[r], __shfl_xor(tm[r], 2, 32));
      tm[r] = fmaxf(tm[r], __shfl_xor(tm[r], 4, 32));
      tm[r] = fmaxf(tm[r], __shfl_xor(tm[r], 8, 32));
    }
#pragma unroll
    for (int r = 0; r < 8; ++r) {
      float mn = fmaxf(m8[r], tm[r]);
      al[r] = __expf(m8[r] - mn);
      m8[r] = mn;
      p0[r] = __expf(s0[r] - mn);
      p1[r] = __expf(s1[r] - mn);
      rs[r] = p0[r] + p1[r];
    }
#pragma unroll
    for (int r = 0; r < 8; ++r) {
      rs[r] += __shfl_xor(rs[r], 1, 32);
      rs[r] += __shfl_xor(rs[r], 2, 32);
      rs[r] += __shfl_xor(rs[r], 4, 32);
      rs[r] += __shfl_xor(rs[r], 8, 32);
      l8[r] = l8[r] * al[r] + rs[r];
    }
#pragma unroll
    for (int nt = 0; nt < 4; ++nt)
#pragma unroll
      for (int r = 0; r < 8; ++r) O[nt][r] *= al[r];

    // C-layout P -> LDS as row-major 16x32 f16 tile (per-wave region).
#pragma unroll
    for (int r = 0; r < 8; ++r) {
      Pl[(row0 + r) * 32 + col]      = (half_t)p0[r];
      Pl[(row0 + r) * 32 + col + 16] = (half_t)p1[r];
    }
    v16h aP = load_A_frag(Pl, 32, lane);  // wave-local; DS ops in order

#pragma unroll
    for (int nt = 0; nt < 4; ++nt) {
#if HAVE_TDM
      v16h bV = load_Bt_frag(Vtl + (size_t)(nt * 16) * 32, 32, lane);
#else
      v16h bV = load_Bt_frag(Vtl + (size_t)(nt * 16) * Sdim + kb, Sdim, lane);
#endif
      O[nt] = WMMA(aP, bV, O[nt]);
    }
#if HAVE_TDM
    __syncthreads();  // all waves done reading cur before it is restaged
#endif
  }

  float inv[8];
#pragma unroll
  for (int r = 0; r < 8; ++r) inv[r] = 1.0f / l8[r];
#pragma unroll
  for (int nt = 0; nt < 4; ++nt)
#pragma unroll
    for (int r = 0; r < 8; ++r) {
      int q = qt * 16 + row0 + r;
      Cc[((size_t)b * Sdim + q) * (Hh * DVc) + h * DVc + nt * 16 + col] =
          (half_t)(O[nt][r] * inv[r]);
    }
}

// ------------------------- output projection -------------------------------
__global__ __launch_bounds__(128) void k_out_gemm(
    const half_t* __restrict__ Cc,   // [B*S, 512]
    const half_t* __restrict__ WoT,  // [512, 512] = Wo^T row-major
    const float* __restrict__ bo,    // [512]
    float* __restrict__ Out) {       // [B*S, 512]
  const int lane = threadIdx.x & 31;
  const int wave = threadIdx.x >> 5;
  const int gw   = blockIdx.x * (blockDim.x >> 5) + wave;
  const int NP = Ddim / 32;
  const int rt = gw / NP;
  const int np = gw % NP;

  const half_t* Ap = Cc + (size_t)rt * 16 * Ddim;
  v8f acc[2] = {};
  for (int ks = 0; ks < Ddim; ks += 32) {
    v16h a = load_A_frag(Ap + ks, Ddim, lane);
#pragma unroll
    for (int nt = 0; nt < 2; ++nt) {
      v16h bf = load_Bt_frag(WoT + (size_t)(np * 32 + nt * 16) * Ddim + ks, Ddim, lane);
      acc[nt] = WMMA(a, bf, acc[nt]);
    }
  }

  const int col  = lane & 15;
  const int row0 = (lane >> 4) * 8;
#pragma unroll
  for (int nt = 0; nt < 2; ++nt)
#pragma unroll
    for (int r = 0; r < 8; ++r) {
      int row  = rt * 16 + row0 + r;
      int colg = np * 32 + nt * 16 + col;
      Out[(size_t)row * Ddim + colg] = acc[nt][r] + bo[colg];
    }
}

// ------------------------------ launcher -----------------------------------

extern "C" void kernel_launch(void* const* d_in, const int* in_sizes, int n_in,
                              void* d_out, int out_size, void* d_ws, size_t ws_size,
                              hipStream_t stream) {
  (void)in_sizes; (void)n_in; (void)out_size; (void)ws_size;

  const float* q_in = (const float*)d_in[0];
  const float* k_in = (const float*)d_in[1];
  const float* v_in = (const float*)d_in[2];
  const float* Wq   = (const float*)d_in[3];
  const float* bq   = (const float*)d_in[4];
  const float* Wk   = (const float*)d_in[5];
  const float* bk   = (const float*)d_in[6];
  const float* Wv   = (const float*)d_in[7];
  const float* bv   = (const float*)d_in[8];
  const float* Wo   = (const float*)d_in[9];
  const float* bo   = (const float*)d_in[10];
  float* out = (float*)d_out;

  const size_t nInput = (size_t)Bdim * Sdim * Ddim;
  const size_t nW     = (size_t)Hh * DKc * Ddim;
  const size_t nWo    = (size_t)Ddim * Ddim;
  const size_t nHead  = (size_t)Bdim * Hh * Sdim * DKc;

  half_t* p = (half_t*)d_ws;
  half_t* qf   = p; p += nInput;
  half_t* kf   = p; p += nInput;
  half_t* vf   = p; p += nInput;
  half_t* Wqt  = p; p += nW;
  half_t* Wkt  = p; p += nW;
  half_t* Wvt  = p; p += nW;
  half_t* WoTt = p; p += nWo;
  half_t* Qh   = p; p += nHead;   // [B,H,S,DK]
  half_t* Kh   = p; p += nHead;   // [B,H,S,DK]
  half_t* Vth  = p; p += nHead;   // [B,H,DV,S]
  half_t* Cc   = p; p += nInput;  // [B,S,H*DV]

  {
    int n = (int)nInput;
    int blocks = (n + 255) / 256;
    k_f32_to_f16<<<blocks, 256, 0, stream>>>(q_in, qf, n);
    k_f32_to_f16<<<blocks, 256, 0, stream>>>(k_in, kf, n);
    k_f32_to_f16<<<blocks, 256, 0, stream>>>(v_in, vf, n);
    int nw = (int)nW;
    k_transpose_to_f16<<<(nw + 255) / 256, 256, 0, stream>>>(Wq, Wqt, Hh, Ddim, DKc);
    k_transpose_to_f16<<<(nw + 255) / 256, 256, 0, stream>>>(Wk, Wkt, Hh, Ddim, DKc);
    k_transpose_to_f16<<<(nw + 255) / 256, 256, 0, stream>>>(Wv, Wvt, Hh, Ddim, DVc);
    int nwo = (int)nWo;
    k_transpose_to_f16<<<(nwo + 255) / 256, 256, 0, stream>>>(Wo, WoTt, 1, Hh * DVc, Ddim);
  }

  {
    const int waves = Bdim * Hh * (Sdim / 16);  // 4096
    k_proj_gemm<<<waves / 4, 128, 0, stream>>>(qf, Wqt, bq, Qh, 0);
    k_proj_gemm<<<waves / 4, 128, 0, stream>>>(kf, Wkt, bk, Kh, 0);
    k_proj_gemm<<<waves / 4, 128, 0, stream>>>(vf, Wvt, bv, Vth, 1);  // V transposed
  }

  {
    const int blocks = Bdim * Hh * (Sdim / 16) / 4;  // 1024, 4 waves share (b,h)
    k_attention<<<blocks, 128, 0, stream>>>(Qh, Kh, Vth, Cc);
  }

  {
    const int waves = (Bdim * Sdim / 16) * (Ddim / 32);  // 8192
    k_out_gemm<<<waves / 4, 128, 0, stream>>>(Cc, WoTt, bo, out);
  }
}
